// GINmodel_57208964382753
// MI455X (gfx1250) — matrix operand order repaired and checked by reading, not compiled, not run on hardware
//
#include <hip/hip_runtime.h>
#include <hip/hip_bf16.h>

// ---------------------------------------------------------------------------
// GINE model for MI455X (gfx1250, wave32, WMMA).
// All big GEMMs use v_wmma_f32_16x16x32_f16 with fp32 accumulation.
// Edge projection is fused with gather(x[src]) + relu + atomic scatter(agg[dst])
// so the 323 MB per-layer edge-message matrix is never written to memory.
// ---------------------------------------------------------------------------

typedef __attribute__((ext_vector_type(16))) _Float16 v16h;
typedef __attribute__((ext_vector_type(8)))  float    v8f;

union FragH {
    v16h     h;
    unsigned u[8];
};

#define N_NODES  50000
#define N_EDGES  800000
#define N_GRAPHS 512
#define DD       101

__device__ __forceinline__ int imin(int a, int b) { return a < b ? a : b; }

// ---------------------------------------------------------------------------
// Fused per-layer edge kernel:
//   e   = edge_attr @ We + be            (WMMA, 128-edge tile per block)
//   m   = relu(x[src] + e)
//   agg[dst] += m                        (fp32 global atomics, L2-resident)
// E = 800000 is an exact multiple of 128 -> no row guards needed.
// ---------------------------------------------------------------------------
__global__ __launch_bounds__(256)
void edge_msg_kernel(const float* __restrict__ edge_attr,
                     const float* __restrict__ We,
                     const float* __restrict__ be,
                     const float* __restrict__ xin,
                     const int*   __restrict__ src,
                     const int*   __restrict__ dst,
                     float*       __restrict__ agg)
{
    __shared__ _Float16 As[128][132];   // edge_attr tile, K padded to 128
    __shared__ _Float16 Ws[112][132];   // We transposed: Ws[n][k]

    const int tid = threadIdx.x;
    const long tileBase = (long)blockIdx.x * 128;

    // Software prefetch of the next tile (streams ahead of the WMMA phase).
    {
        long nextBase = (tileBase + 128) * DD;          // floats
        if (blockIdx.x + 1 < N_EDGES / 128) {
            // one byte per lane pulls whole cachelines; spread 256 lanes over tile
            const char* p = (const char*)(edge_attr + nextBase) + (long)tid * 202;
            __builtin_prefetch(p, 0, 1);
        }
    }

    // Stage W transposed (f32 -> f16), zero-padded to 112 x 128.
    for (int idx = tid; idx < 112 * 128; idx += 256) {
        int n = idx % 112, k = idx / 112;
        float w = (n < DD && k < DD) ? We[k * DD + n] : 0.f;
        Ws[n][k] = (_Float16)w;
    }
    // Stage A tile (streaming data: non-temporal so L2 keeps node arrays).
    for (int idx = tid; idx < 128 * 128; idx += 256) {
        int r = idx >> 7, k = idx & 127;
        float v = 0.f;
        if (k < DD)
            v = __builtin_nontemporal_load(&edge_attr[(tileBase + r) * DD + k]);
        As[r][k] = (_Float16)v;
    }
    __syncthreads();

    const int wave  = tid >> 5;
    const int lane  = tid & 31;
    const int rbase = wave * 16;         // this wave's 16 rows of the tile
    const int mcol  = lane & 15;
    const int hi    = lane >> 4;
    const int mlo   = hi * 8;            // C-fragment row offset

    // A fragments for the 4 k-steps (K = 0..127).
    FragH a[4];
#pragma unroll
    for (int kk = 0; kk < 4; ++kk) {
        const int row = rbase + mcol;
        const int kb  = 32 * kk + hi * 8;
        const unsigned* p0 = (const unsigned*)&As[row][kb];
        const unsigned* p1 = (const unsigned*)&As[row][kb + 16];
#pragma unroll
        for (int i = 0; i < 4; ++i) { a[kk].u[i] = p0[i]; a[kk].u[4 + i] = p1[i]; }
    }

    // src/dst for this lane's 8 output rows (same across all N tiles).
    int sv[8], dv[8];
#pragma unroll
    for (int r = 0; r < 8; ++r) {
        long e = tileBase + rbase + mlo + r;
        sv[r] = src[e];
        dv[r] = dst[e];
    }

    const int kb0 = hi * 8;              // B-fragment K base
    for (int nt = 0; nt < 7; ++nt) {     // 7 tiles cover 112 >= 101 columns
        const int ncol = nt * 16 + mcol;

        // Preload ALL four B fragments -> one dscnt wait, 4 back-to-back WMMAs.
        FragH b[4];
#pragma unroll
        for (int kk = 0; kk < 4; ++kk) {
            const unsigned* p = (const unsigned*)&Ws[ncol][32 * kk + kb0];
#pragma unroll
            for (int i = 0; i < 4; ++i) { b[kk].u[i] = p[i]; b[kk].u[4 + i] = p[8 + i]; }
        }
        v8f c = {};
#pragma unroll
        for (int kk = 0; kk < 4; ++kk)
            c = __builtin_amdgcn_wmma_f32_16x16x32_f16(
                    false, a[kk].h, false, b[kk].h, (short)0, c, false, false);

        if (ncol < DD) {                 // guard only the epilogue
            const float bias = be[ncol];
            // Phase 1: issue all 8 gathers (kept in flight together).
            float xv[8];
#pragma unroll
            for (int r = 0; r < 8; ++r)
                xv[r] = xin[(long)sv[r] * DD + ncol];
            // Phase 2: compute + scatter atomics.
#pragma unroll
            for (int r = 0; r < 8; ++r) {
                float v = fmaxf(c[r] + bias + xv[r], 0.f);
                atomicAdd(&agg[(long)dv[r] * DD + ncol], v);
            }
        }
    }
}

// ---------------------------------------------------------------------------
// Generic WMMA GEMM: C[M,N] = act( (A (+A2)) @ W + bias )
// K <= 128 (padded), N arbitrary (processed in 112-wide chunks).
// ---------------------------------------------------------------------------
__global__ __launch_bounds__(256)
void gemm_bias_kernel(const float* __restrict__ A,
                      const float* __restrict__ A2,   // optional second addend
                      const float* __restrict__ W,
                      const float* __restrict__ bias,
                      float*       __restrict__ C,
                      int M, int K, int N, int doRelu)
{
    __shared__ _Float16 As[128][132];
    __shared__ _Float16 Ws[112][132];

    const int tid = threadIdx.x;
    const long rowBase = (long)blockIdx.x * 128;

    for (int idx = tid; idx < 128 * 128; idx += 256) {
        int r = idx >> 7, k = idx & 127;
        long row = rowBase + r;
        float v = 0.f;
        if (k < K && row < M) {
            v = A[row * K + k];
            if (A2) v += A2[row * K + k];
        }
        As[r][k] = (_Float16)v;
    }
    __syncthreads();

    const int wave  = tid >> 5;
    const int lane  = tid & 31;
    const int rbase = wave * 16;
    const int mcol  = lane & 15;
    const int hi    = lane >> 4;
    const int mlo   = hi * 8;

    FragH a[4];
#pragma unroll
    for (int kk = 0; kk < 4; ++kk) {
        const int row = rbase + mcol;
        const int kb  = 32 * kk + hi * 8;
        const unsigned* p0 = (const unsigned*)&As[row][kb];
        const unsigned* p1 = (const unsigned*)&As[row][kb + 16];
#pragma unroll
        for (int i = 0; i < 4; ++i) { a[kk].u[i] = p0[i]; a[kk].u[4 + i] = p1[i]; }
    }

    const int kb0 = hi * 8;
    for (int chunk = 0; chunk < N; chunk += 112) {
        __syncthreads();                 // previous chunk's Ws reads done
        for (int idx = tid; idx < 112 * 128; idx += 256) {
            int n = idx % 112, k = idx / 112;
            int gn = chunk + n;
            float w = (gn < N && k < K) ? W[k * N + gn] : 0.f;
            Ws[n][k] = (_Float16)w;
        }
        __syncthreads();

        const int ntmax = imin(7, (N - chunk + 15) >> 4);
        for (int nt = 0; nt < ntmax; ++nt) {
            const int ncol = nt * 16 + mcol;

            FragH b[4];
#pragma unroll
            for (int kk = 0; kk < 4; ++kk) {
                const unsigned* p = (const unsigned*)&Ws[ncol][32 * kk + kb0];
#pragma unroll
                for (int i = 0; i < 4; ++i) { b[kk].u[i] = p[i]; b[kk].u[4 + i] = p[8 + i]; }
            }
            v8f c = {};
#pragma unroll
            for (int kk = 0; kk < 4; ++kk)
                c = __builtin_amdgcn_wmma_f32_16x16x32_f16(
                        false, a[kk].h, false, b[kk].h, (short)0, c, false, false);

            const int gcol = chunk + nt * 16 + mcol;
            if (gcol < N) {
                const float bs = bias[gcol];
#pragma unroll
                for (int r = 0; r < 8; ++r) {
                    long row = rowBase + rbase + mlo + r;
                    if (row < M) {
                        float v = c[r] + bs;
                        if (doRelu) v = fmaxf(v, 0.f);
                        C[row * N + gcol] = v;
                    }
                }
            }
        }
    }
}

// ---------------------------------------------------------------------------
__global__ void zero_kernel(float* __restrict__ p, long n)
{
    long i = (long)blockIdx.x * blockDim.x + threadIdx.x;
    if (i < n) p[i] = 0.f;
}

// global_add_pool: g[batch[n], c] += h2[n, c]   (g is 512x200, L2-resident)
// One block per node row: no integer division, coalesced h2 reads.
__global__ __launch_bounds__(256)
void pool_kernel(const float* __restrict__ h2,
                 const int*   __restrict__ batch,
                 float*       __restrict__ g)
{
    const int n = blockIdx.x;
    const int c = threadIdx.x;
    if (c < 200) {
        const int b = batch[n];
        atomicAdd(&g[(long)b * 200 + c], h2[(long)n * 200 + c]);
    }
}

// Tiny head: out[gi] = | relu(g[gi] @ Wl + bl) @ Wl2 + bl2 |
__global__ __launch_bounds__(64)
void head_kernel(const float* __restrict__ g,
                 const float* __restrict__ Wl,  const float* __restrict__ bl,
                 const float* __restrict__ Wl2, const float* __restrict__ bl2,
                 float* __restrict__ out)
{
    __shared__ float t[50];
    const int gi = blockIdx.x;
    const int j  = threadIdx.x;
    if (j < 50) {
        float acc = bl[j];
        for (int k = 0; k < 200; ++k)
            acc += g[(long)gi * 200 + k] * Wl[k * 50 + j];
        t[j] = fmaxf(acc, 0.f);
    }
    __syncthreads();
    if (j == 0) {
        float s = bl2[0];
        for (int k = 0; k < 50; ++k) s += t[k] * Wl2[k];
        out[gi] = fabsf(s);
    }
}

// ---------------------------------------------------------------------------
extern "C" void kernel_launch(void* const* d_in, const int* in_sizes, int n_in,
                              void* d_out, int out_size, void* d_ws, size_t ws_size,
                              hipStream_t stream)
{
    (void)in_sizes; (void)n_in; (void)out_size; (void)ws_size;

    const float* x         = (const float*)d_in[0];
    const int*   edge_idx  = (const int*)  d_in[1];
    // d_in[2] = edge_type (unused by the reference computation)
    const float* edge_attr = (const float*)d_in[3];
    const int*   batch     = (const int*)  d_in[4];
    const float* We1 = (const float*)d_in[5];
    const float* be1 = (const float*)d_in[6];
    const float* W1a = (const float*)d_in[7];
    const float* b1a = (const float*)d_in[8];
    const float* W1b = (const float*)d_in[9];
    const float* b1b = (const float*)d_in[10];
    const float* We2 = (const float*)d_in[11];
    const float* be2 = (const float*)d_in[12];
    const float* W2a = (const float*)d_in[13];
    const float* b2a = (const float*)d_in[14];
    const float* W2b = (const float*)d_in[15];
    const float* b2b = (const float*)d_in[16];
    const float* Wl  = (const float*)d_in[17];
    const float* bl  = (const float*)d_in[18];
    const float* Wl2 = (const float*)d_in[19];
    const float* bl2 = (const float*)d_in[20];
    float* out = (float*)d_out;

    const int* src = edge_idx;
    const int* dst = edge_idx + N_EDGES;

    // Workspace layout (floats).
    float* ws   = (float*)d_ws;
    const long ND = (long)N_NODES * DD;              // 5,050,000
    float* agg = ws;                                 // [50000,101] (both layers)
    float* h1  = agg + ND;                           // [50000,101]
    float* u   = h1  + ND;                           // [50000,101] / [50000,100]
    float* h2  = u   + ND;                           // [50000,200]
    float* g   = h2  + (long)N_NODES * 200;          // [512,200]

    const int TB = 256;
    const int edgeBlocks = N_EDGES / 128;            // 6250, exact
    const int nodeBlocks = (N_NODES + 127) / 128;    // 391

    // ---- Layer 1 ----
    zero_kernel<<<(int)((ND + TB - 1) / TB), TB, 0, stream>>>(agg, ND);
    edge_msg_kernel<<<edgeBlocks, TB, 0, stream>>>(edge_attr, We1, be1, x, src, dst, agg);
    gemm_bias_kernel<<<nodeBlocks, TB, 0, stream>>>(x,  agg, W1a, b1a, u,  N_NODES, DD, DD, 1);
    gemm_bias_kernel<<<nodeBlocks, TB, 0, stream>>>(u, nullptr, W1b, b1b, h1, N_NODES, DD, DD, 1); // outer relu

    // ---- Layer 2 ----
    zero_kernel<<<(int)((ND + TB - 1) / TB), TB, 0, stream>>>(agg, ND);
    edge_msg_kernel<<<edgeBlocks, TB, 0, stream>>>(edge_attr, We2, be2, h1, src, dst, agg);
    gemm_bias_kernel<<<nodeBlocks, TB, 0, stream>>>(h1, agg, W2a, b2a, u,  N_NODES, DD, 100, 1);
    gemm_bias_kernel<<<nodeBlocks, TB, 0, stream>>>(u, nullptr, W2b, b2b, h2, N_NODES, 100, 200, 1); // outer relu

    // ---- Pool + head ----
    const long GN = (long)N_GRAPHS * 200;
    zero_kernel<<<(int)((GN + TB - 1) / TB), TB, 0, stream>>>(g, GN);
    pool_kernel<<<N_NODES, TB, 0, stream>>>(h2, batch, g);
    head_kernel<<<N_GRAPHS, 64, 0, stream>>>(g, Wl, bl, Wl2, bl2, out);
}
